// BasicTransBlock_15676630631098
// MI455X (gfx1250) — compile-verified, matrix operand order checked by reading, not run on hardware
//
#include <hip/hip_runtime.h>
#include <hip/hip_bf16.h>

typedef _Float16 v8h  __attribute__((ext_vector_type(8)));
typedef _Float16 v16h __attribute__((ext_vector_type(16)));
typedef float    v8f  __attribute__((ext_vector_type(8)));

#define DEVINL __device__ __forceinline__

constexpr int   B_    = 128;
constexpr int   C_    = 512;
constexpr int   HS    = 14;
constexpr int   HW    = 196;
constexpr int   HEADS = 4;
constexpr int   DHD   = 128;
constexpr int   M1    = B_ * HW;          // 25088
constexpr float EPS   = 1e-5f;
constexpr float SCALE = 0.08838834764831845f;  // 128^-0.5

// ---------------------------------------------------------------------------
// WMMA helpers.  16-bit A/B fragment layout (ISA 7.12.2): lane = {row|col}&15,
// lane>>4 selects K-half; VGPRs 0-3 hold K = kh*8 .. kh*8+7 (one b128 load),
// VGPRs 4-7 hold K = 16+kh*8 .. (second b128 load).
// ---------------------------------------------------------------------------
DEVINL v16h load_frag(const _Float16* base, int ld, int lane, int kk) {
  const int row = lane & 15;
  const int kh  = (lane >> 4) & 1;
  const _Float16* p = base + row * ld + kk + kh * 8;
  v8h lo = *(const v8h*)(p);
  v8h hi = *(const v8h*)(p + 16);
  v16h f;
#pragma unroll
  for (int i = 0; i < 8; ++i) { f[i] = lo[i]; f[i + 8] = hi[i]; }
  return f;
}

DEVINL v8f wmma16(v16h a, v16h b, v8f c) {
  return __builtin_amdgcn_wmma_f32_16x16x32_f16(false, a, false, b,
                                                (short)0, c, false, false);
}

DEVINL v8f vzero8() { v8f a;
#pragma unroll
  for (int i = 0; i < 8; ++i) a[i] = 0.f; return a; }

// Generic shared pointer -> raw LDS byte offset (flat aperture keeps the LDS
// address in bits [31:0], ISA 10.2).
DEVINL unsigned lds_addr(const void* p) { return (unsigned)(uintptr_t)p; }

// CDNA5 async memory->LDS copy, 16B per lane, tracked by ASYNCcnt.
DEVINL void async_copy_b128(unsigned lds, const void* gptr) {
  asm volatile("global_load_async_to_lds_b128 %0, %1, off"
               :: "v"(lds), "v"((unsigned long long)(uintptr_t)gptr)
               : "memory");
}
DEVINL void wait_async0() {
  asm volatile("s_wait_asynccnt 0" ::: "memory");
}

// ---------------------------------------------------------------------------
// Kernel: f32 -> f16 convert (weight repack; [o][c] is already B-matrix [N][K])
// ---------------------------------------------------------------------------
__global__ void cvt_f16_kernel(const float* __restrict__ src,
                               _Float16* __restrict__ dst, int n) {
  int i = blockIdx.x * 256 + threadIdx.x;
  if (i < n) dst[i] = (_Float16)src[i];
}

// ---------------------------------------------------------------------------
// Kernel: (optional BN) + depthwise 3x3 -> f16 A-matrix [B*HW][C]
// ---------------------------------------------------------------------------
template <bool BN>
__global__ void dw3x3_kernel(const float* __restrict__ src,   // NCHW f32
                             const float* __restrict__ wdw,   // [C][9]
                             const float* __restrict__ g, const float* __restrict__ bt,
                             const float* __restrict__ mu, const float* __restrict__ var,
                             _Float16* __restrict__ out) {    // [B*HW][C]
  int idx = blockIdx.x * 256 + threadIdx.x;
  if (idx >= B_ * C_ * HW) return;
  int p = idx % HW;
  int c = (idx / HW) % C_;
  int b = idx / (HW * C_);
  int py = p / HS, px = p % HS;
  float inv = 1.f, beta = 0.f;
  if (BN) { inv = g[c] * rsqrtf(var[c] + EPS); beta = bt[c] - mu[c] * inv; }
  const float* sp = src + ((size_t)b * C_ + c) * HW;
  const float* wp = wdw + c * 9;
  float acc = 0.f;
#pragma unroll
  for (int ky = 0; ky < 3; ++ky) {
#pragma unroll
    for (int kx = 0; kx < 3; ++kx) {
      int y = py + ky - 1, x = px + kx - 1;
      if (y >= 0 && y < HS && x >= 0 && x < HS) {
        float xv = sp[y * HS + x];
        if (BN) xv = xv * inv + beta;
        acc += xv * wp[ky * 3 + kx];
      }
    }
  }
  out[((size_t)b * HW + p) * C_ + c] = (_Float16)acc;
}

// ---------------------------------------------------------------------------
// WMMA GEMM: D[M][N] = A[M][K] * B[N][K]^T, 128x128 macro tile, KC=64 chunks.
// All launches use M % 128 == 0, N % 128 == 0, K % 64 == 0, so LDS staging is
// unguarded async copies (GLOBAL_LOAD_ASYNC_TO_LDS_B128 / ASYNCcnt).
// 8 waves: wave w owns rows [16w,16w+16) x all 128 N (A-fragment reuse x8).
// EPI 0: scatter qkv -> Q head-major f16 + KV staging f16
// EPI 1: +x residual, store residual f32 NCHW, BN2+ReLU -> f16 A-matrix
// EPI 2: +residual -> d_out f32 NCHW
// ---------------------------------------------------------------------------
constexpr int TM = 128, TN = 128, KC = 64, LDT = KC + 8;   // +8 halves de-banks

template <int EPI>
__global__ __launch_bounds__(256) void gemm_wmma_kernel(
    const _Float16* __restrict__ A, const _Float16* __restrict__ Bm,
    int M, int N, int K,
    _Float16* __restrict__ q_out, _Float16* __restrict__ kv_out,       // EPI 0
    const float* __restrict__ xres, float* __restrict__ res_out,        // EPI 1
    _Float16* __restrict__ y_next,
    const float* __restrict__ g, const float* __restrict__ bt,
    const float* __restrict__ mu, const float* __restrict__ var,
    const float* __restrict__ res_in, float* __restrict__ final_out) {  // EPI 2
  __shared__ _Float16 As[TM][LDT];
  __shared__ _Float16 Bs[TN][LDT];
  const int tid  = threadIdx.x;
  const int wave = tid >> 5, lane = tid & 31;
  const int m0 = blockIdx.x * TM, n0 = blockIdx.y * TN;

  v8f acc[8];
#pragma unroll
  for (int t = 0; t < 8; ++t) acc[t] = vzero8();

  for (int k0 = 0; k0 < K; k0 += KC) {
    // async memory->LDS staging of the A and B K-chunks (no VGPR bounce)
#pragma unroll
    for (int i = 0; i < 4; ++i) {               // TM*(KC/8)/256 == 4
      int id = tid + i * 256;
      int r = id >> 3, s = id & 7;
      async_copy_b128(lds_addr(&As[r][s * 8]),
                      A + (size_t)(m0 + r) * K + k0 + s * 8);
    }
#pragma unroll
    for (int i = 0; i < 4; ++i) {
      int id = tid + i * 256;
      int r = id >> 3, s = id & 7;
      async_copy_b128(lds_addr(&Bs[r][s * 8]),
                      Bm + (size_t)(n0 + r) * K + k0 + s * 8);
    }
    wait_async0();
    __syncthreads();
#pragma unroll
    for (int kk = 0; kk < KC; kk += 32) {
      v16h af = load_frag(&As[wave * 16][0], LDT, lane, kk);
#pragma unroll
      for (int t = 0; t < 8; ++t) {
        v16h bf = load_frag(&Bs[t * 16][0], LDT, lane, kk);
        acc[t] = wmma16(af, bf, acc[t]);
      }
    }
    __syncthreads();
  }

  // Epilogue. C/D layout: lane gives N=lane&15, VGPR r gives M=r+8*(lane>>4).
  const int rg = lane >> 4, ln = lane & 15;
#pragma unroll
  for (int t = 0; t < 8; ++t) {
    int gn = n0 + t * 16 + ln;
#pragma unroll
    for (int r = 0; r < 8; ++r) {
      int gm = m0 + wave * 16 + rg * 8 + r;
      if (gm >= M || gn >= N) continue;
      float v = acc[t][r];
      if constexpr (EPI == 0) {
        int b = gm / HW, p = gm % HW;
        if (gn < C_) {                       // Q: c = d*HEADS + head
          int head = gn & 3, d = gn >> 2;
          q_out[(((size_t)b * HEADS + head) * HW + p) * DHD + d] = (_Float16)v;
        } else {
          kv_out[((size_t)b * HW + p) * 1024 + (gn - C_)] = (_Float16)v;
        }
      } else if constexpr (EPI == 1) {
        int b = gm / HW, p = gm % HW, c = gn;
        size_t xi = ((size_t)b * C_ + c) * HW + p;
        float rsd = v + xres[xi];
        res_out[xi] = rsd;
        float inv = g[c] * rsqrtf(var[c] + EPS);
        float bn  = rsd * inv + (bt[c] - mu[c] * inv);
        y_next[(size_t)gm * C_ + gn] = (_Float16)fmaxf(bn, 0.f);
      } else {
        size_t xi = ((size_t)(gm / HW) * C_ + gn) * HW + (gm % HW);
        final_out[xi] = v + res_in[xi];
      }
    }
  }
}

// ---------------------------------------------------------------------------
// Bilinear 14x14 -> 7x7 (align_corners), scatter to head-major f16
// ---------------------------------------------------------------------------
__global__ void downsample_kernel(const _Float16* __restrict__ kv,
                                  _Float16* __restrict__ Kh,
                                  _Float16* __restrict__ Vh) {
  int idx = blockIdx.x * 256 + threadIdx.x;
  if (idx >= B_ * 2 * C_ * 49) return;
  int rp = idx % 49;
  int c  = (idx / 49) % C_;
  int sel = (idx / (49 * C_)) & 1;
  int b  = idx / (49 * C_ * 2);
  int ry = rp / 7, rx = rp % 7;
  float sy = (float)ry * (13.f / 6.f), sx = (float)rx * (13.f / 6.f);
  int y0 = (int)floorf(sy), x0 = (int)floorf(sx);
  int y1 = min(y0 + 1, HS - 1), x1 = min(x0 + 1, HS - 1);
  float wy = sy - (float)y0, wx = sx - (float)x0;
  const _Float16* base = kv + (size_t)b * HW * 1024 + sel * C_ + c;
  auto at = [&](int y, int x) { return (float)base[(size_t)(y * HS + x) * 1024]; };
  float top = at(y0, x0) * (1.f - wx) + at(y0, x1) * wx;
  float bot = at(y1, x0) * (1.f - wx) + at(y1, x1) * wx;
  float v   = top * (1.f - wy) + bot * wy;
  int head = c & 3, d = c >> 2;
  _Float16* dst = sel ? Vh : Kh;
  dst[(((size_t)b * HEADS + head) * 49 + rp) * DHD + d] = (_Float16)v;
}

// ---------------------------------------------------------------------------
// Attention: one workgroup per (b, head).  LDS-resident Q/K/V/P (~172 KB of
// the 320 KB WGP LDS).  Q/K tiles staged with async copies; rel-pos bias
// slice for this head cached in LDS (kills the per-element global gather).
// ---------------------------------------------------------------------------
constexpr int LQ = 136;  // 128 + 8 halves
constexpr int LP = 72;   // 64 + 8 halves

__global__ __launch_bounds__(256) void attn_kernel(
    const _Float16* __restrict__ Qh, const _Float16* __restrict__ Kh,
    const _Float16* __restrict__ Vh, const float* __restrict__ rel_table,
    float* __restrict__ Oattn) {
  __shared__ _Float16 Qs[208 * LQ];
  __shared__ _Float16 Ks[64 * LQ];
  __shared__ _Float16 Vs[128 * LP];   // transposed: [d][j]
  __shared__ float    Sf[208 * 64];
  __shared__ _Float16 Ps[208 * LP];
  __shared__ float    RelS[169];      // rel_table[:, head]

  const int bh = blockIdx.x, head = bh & 3, b = bh >> 2;
  const int tid = threadIdx.x, wave = tid >> 5, lane = tid & 31;
  const _Float16* Qg = Qh + (size_t)bh * HW * DHD;
  const _Float16* Kg = Kh + (size_t)bh * 49 * DHD;
  const _Float16* Vg = Vh + (size_t)bh * 49 * DHD;

  // zero-fill pad rows with plain ds stores
  {
    v8h z;
#pragma unroll
    for (int i = 0; i < 8; ++i) z[i] = (_Float16)0.f;
    for (int i = tid; i < 12 * 16; i += 256) {        // Q rows 196..207
      int r = 196 + (i >> 4), s = i & 15;
      *(v8h*)&Qs[r * LQ + s * 8] = z;
    }
    for (int i = tid; i < 15 * 16; i += 256) {        // K rows 49..63
      int r = 49 + (i >> 4), s = i & 15;
      *(v8h*)&Ks[r * LQ + s * 8] = z;
    }
  }
  // async memory->LDS for the live Q/K rows
  for (int i = tid; i < HW * 16; i += 256) {
    int r = i >> 4, s = i & 15;
    async_copy_b128(lds_addr(&Qs[r * LQ + s * 8]), Qg + (size_t)r * DHD + s * 8);
  }
  for (int i = tid; i < 49 * 16; i += 256) {
    int r = i >> 4, s = i & 15;
    async_copy_b128(lds_addr(&Ks[r * LQ + s * 8]), Kg + (size_t)r * DHD + s * 8);
  }
  // V transpose [d][j] (gather; VGPR-mediated)
  for (int i = tid; i < 128 * 64; i += 256) {
    int d = i >> 6, j = i & 63;
    _Float16 v = (_Float16)0.f;
    if (j < 49) v = Vg[(size_t)j * DHD + d];
    Vs[d * LP + j] = v;
  }
  // rel-pos bias slice for this head
  if (tid < 169) RelS[tid] = rel_table[tid * HEADS + head];
  wait_async0();
  __syncthreads();

  // ---- S = Q K^T : 13 M-tiles x 4 N-tiles, wave-cyclic --------------------
  const int rg = lane >> 4, ln = lane & 15;
  for (int t = wave; t < 52; t += 8) {
    int mt = t >> 2, nt = t & 3;
    v8f acc = vzero8();
#pragma unroll
    for (int kk = 0; kk < 128; kk += 32) {
      v16h a = load_frag(&Qs[mt * 16 * LQ], LQ, lane, kk);
      v16h k = load_frag(&Ks[nt * 16 * LQ], LQ, lane, kk);
      acc = wmma16(a, k, acc);
    }
#pragma unroll
    for (int r = 0; r < 8; ++r) {
      int i = mt * 16 + rg * 8 + r;   // query pos (padded)
      int j = nt * 16 + ln;           // key pos (padded)
      float s;
      if (i < HW && j < 49) {
        int py = i / HS, px = i % HS;
        int iy = py >> 1, ix = px >> 1;
        int jy = j / 7,  jx = j % 7;
        int rel = (iy - jy + 6) * 13 + (ix - jx + 6);
        s = (acc[r] + RelS[rel]) * SCALE;
      } else {
        s = -1e30f;                   // mask padding out of the softmax
      }
      Sf[i * 64 + j] = s;
    }
  }
  __syncthreads();

  // ---- softmax over j (49), write P f16 with zero padding -----------------
  for (int r = tid; r < 208; r += 256) {
    if (r < HW) {
      float mx = -1e30f;
      for (int j = 0; j < 49; ++j) mx = fmaxf(mx, Sf[r * 64 + j]);
      float sum = 0.f;
      for (int j = 0; j < 49; ++j) sum += __expf(Sf[r * 64 + j] - mx);
      float rs = 1.f / sum;
      for (int j = 0; j < 49; ++j)
        Ps[r * LP + j] = (_Float16)(__expf(Sf[r * 64 + j] - mx) * rs);
      for (int j = 49; j < 64; ++j) Ps[r * LP + j] = (_Float16)0.f;
    } else {
      for (int j = 0; j < 64; ++j) Ps[r * LP + j] = (_Float16)0.f;
    }
  }
  __syncthreads();

  // ---- O = P V : 13 M-tiles x 8 N-tiles, wave-cyclic ----------------------
  for (int t = wave; t < 104; t += 8) {
    int mt = t >> 3, nt = t & 7;
    v8f acc = vzero8();
#pragma unroll
    for (int kk = 0; kk < 64; kk += 32) {
      v16h a = load_frag(&Ps[mt * 16 * LP], LP, lane, kk);
      v16h v = load_frag(&Vs[nt * 16 * LP], LP, lane, kk);
      acc = wmma16(a, v, acc);
    }
#pragma unroll
    for (int r = 0; r < 8; ++r) {
      int p = mt * 16 + rg * 8 + r;
      int d = nt * 16 + ln;
      if (p < HW)
        Oattn[((size_t)b * C_ + (d * HEADS + head)) * HW + p] = acc[r];
    }
  }
}

// ---------------------------------------------------------------------------
extern "C" void kernel_launch(void* const* d_in, const int* in_sizes, int n_in,
                              void* d_out, int out_size, void* d_ws, size_t ws_size,
                              hipStream_t stream) {
  const float* x        = (const float*)d_in[0];
  const float* bn1_g    = (const float*)d_in[1];
  const float* bn1_b    = (const float*)d_in[2];
  const float* bn1_m    = (const float*)d_in[3];
  const float* bn1_v    = (const float*)d_in[4];
  const float* w_qkv_dw = (const float*)d_in[5];
  const float* w_qkv_pw = (const float*)d_in[6];
  const float* rel_tab  = (const float*)d_in[7];
  const float* w_out_dw = (const float*)d_in[8];
  const float* w_out_pw = (const float*)d_in[9];
  const float* bn2_g    = (const float*)d_in[10];
  const float* bn2_b    = (const float*)d_in[11];
  const float* bn2_m    = (const float*)d_in[12];
  const float* bn2_v    = (const float*)d_in[13];
  const float* w_mlp    = (const float*)d_in[14];
  float* out = (float*)d_out;

  char* w = (char*)d_ws;
  _Float16* wqkvh = (_Float16*)w; w += (size_t)3 * C_ * C_ * 2;
  _Float16* wouth = (_Float16*)w; w += (size_t)C_ * C_ * 2;
  _Float16* wmlph = (_Float16*)w; w += (size_t)C_ * C_ * 2;
  _Float16* y1    = (_Float16*)w; w += (size_t)M1 * C_ * 2;     // reused as y2
  _Float16* y3    = (_Float16*)w; w += (size_t)M1 * C_ * 2;
  _Float16* Qh    = (_Float16*)w; w += (size_t)B_ * HEADS * HW * DHD * 2;
  _Float16* KVraw = (_Float16*)w; w += (size_t)B_ * HW * 1024 * 2;
  _Float16* Kh    = (_Float16*)w; w += (size_t)B_ * HEADS * 49 * DHD * 2;
  _Float16* Vh    = (_Float16*)w; w += (size_t)B_ * HEADS * 49 * DHD * 2;
  float*    res   = (float*)w;    w += (size_t)B_ * C_ * HW * 4;
  float*    Oattn = (float*)KVraw;  // KVraw dead after downsample; safe alias

  // 1) weight repack f32 -> f16 (already B-matrix [N][K] layout)
  cvt_f16_kernel<<<(3 * C_ * C_ + 255) / 256, 256, 0, stream>>>(w_qkv_pw, wqkvh, 3 * C_ * C_);
  cvt_f16_kernel<<<(C_ * C_ + 255) / 256, 256, 0, stream>>>(w_out_pw, wouth, C_ * C_);
  cvt_f16_kernel<<<(C_ * C_ + 255) / 256, 256, 0, stream>>>(w_mlp, wmlph, C_ * C_);

  // 2) BN1 + depthwise 3x3 -> y1 f16 [25088][512]
  dw3x3_kernel<true><<<(B_ * C_ * HW + 255) / 256, 256, 0, stream>>>(
      x, w_qkv_dw, bn1_g, bn1_b, bn1_m, bn1_v, y1);

  // 3) qkv GEMM [25088 x 1536] with scatter epilogue
  gemm_wmma_kernel<0><<<dim3(M1 / TM, (3 * C_) / TN), 256, 0, stream>>>(
      y1, wqkvh, M1, 3 * C_, C_, Qh, KVraw,
      nullptr, nullptr, nullptr, nullptr, nullptr, nullptr, nullptr,
      nullptr, nullptr);

  // 4) bilinear 14->7 downsample of K,V into head-major f16
  downsample_kernel<<<(B_ * 2 * C_ * 49 + 255) / 256, 256, 0, stream>>>(KVraw, Kh, Vh);

  // 5) attention (one workgroup per (b,head))
  attn_kernel<<<B_ * HEADS, 256, 0, stream>>>(Qh, Kh, Vh, rel_tab, Oattn);

  // 6) depthwise 3x3 on attention output -> y2 (reuse y1) f16
  dw3x3_kernel<false><<<(B_ * C_ * HW + 255) / 256, 256, 0, stream>>>(
      Oattn, w_out_dw, nullptr, nullptr, nullptr, nullptr, y1);

  // 7) out-proj GEMM + residual + BN2/ReLU -> res f32, y3 f16
  gemm_wmma_kernel<1><<<dim3(M1 / TM, C_ / TN), 256, 0, stream>>>(
      y1, wouth, M1, C_, C_, nullptr, nullptr,
      x, res, y3, bn2_g, bn2_b, bn2_m, bn2_v, nullptr, nullptr);

  // 8) MLP GEMM + residual -> d_out
  gemm_wmma_kernel<2><<<dim3(M1 / TM, C_ / TN), 256, 0, stream>>>(
      y3, wmlph, M1, C_, C_, nullptr, nullptr,
      nullptr, nullptr, nullptr, nullptr, nullptr, nullptr, nullptr,
      res, out);

  (void)in_sizes; (void)n_in; (void)out_size; (void)ws_size;
}
